// ActorNetwork_74594991997205
// MI455X (gfx1250) — matrix operand order repaired and checked by reading, not compiled, not run on hardware
//
#include <hip/hip_runtime.h>
#include <hip/hip_bf16.h>

// ---------------------------------------------------------------------------
// CDNA5 (gfx1250) ActorNetwork forward pass.
// Every MLP layer has K<=29, N<=16  =>  one v_wmma_f32_16x16x32_f16 per
// 16-row tile per layer (wave32: one wave owns a tile). Weights are
// pre-converted once into zero-padded f16 B-layout tiles [32x16]
// (column-major col*32+k) so a lane's B fragment is two contiguous 16B loads,
// hoisted out of the tile loop. Activations stage through per-wave LDS tiles
// in the ISA 16-bit A layout (row-major [16][32] halves). Tile-based kernels
// use grid-stride loops (uniform iteration count per block) to amortize
// LDS zero-init / weight loads / launch overhead. Segment sums use hardware
// f32 global atomics. Whole working set (~180 MB) is L2-resident (192 MB).
// ---------------------------------------------------------------------------

typedef _Float16 half8_t  __attribute__((ext_vector_type(8)));
typedef _Float16 half16_t __attribute__((ext_vector_type(16)));
typedef float    float8_t __attribute__((ext_vector_type(8)));

#define LEAKY_SLOPE 0.01f
#define TPB 256
#define WPB 8           // waves per block
#define MAXBLK 1024

// 18 layers total, in d_in order (W at d_in[5+2j], b at d_in[6+2j]):
//  0,1   node_prep   (5->16 act, 16->8)
//  2,3   node_msg    (8->16 act, 16->8)
//  4,5   node_update (8->16 act, 16->8)
//  6,7,8 dag_msg     (13->16 act, 16->8 act, 8->8)
//  9..11 glob_msg    (8->16 act, 16->8 act, 8->8)
// 12..14 node_score  (29->16 act, 16->8 act, 8->1)
// 15..17 dag_score   (20->16 act, 16->8 act, 8->1)
__constant__ int KDIM[18] = {5,16, 8,16, 8,16, 13,16,8, 8,16,8, 29,16,8, 20,16,8};
__constant__ int NDIM[18] = {16,8, 16,8, 16,8, 16,8,8, 16,8,8, 16,8,1, 16,8,1};

struct Params { const float* W[18]; const float* B[18]; };

// ---- WMMA fragment helpers -------------------------------------------------

// A fragment: 16x32 f16 tile, LDS row-major [row][k] (row stride 32 halves).
// Lane l: row = l&15, kb = (l>>4)*8; elems 0..7 = K kb..kb+7, 8..15 = kb+16..kb+23.
__device__ inline half16_t load_A_lds(const _Float16* A, int lane) {
  const int row = lane & 15;
  const int kb  = (lane >> 4) << 3;
  half8_t lo = *(const half8_t*)(A + row * 32 + kb);
  half8_t hi = *(const half8_t*)(A + row * 32 + kb + 16);
  return __builtin_shufflevector(lo, hi, 0,1,2,3,4,5,6,7,8,9,10,11,12,13,14,15);
}

// B fragment: 32x16 f16 tile stored column-major in ws (col*32 + k).
// Lane l: col = l&15, kb = (l>>4)*8; elems 0..7 = K kb..kb+7, 8..15 = kb+16..kb+23.
__device__ inline half16_t load_B(const _Float16* wt, int lane) {
  const int col = lane & 15;
  const int kb  = (lane >> 4) << 3;
  half8_t lo = *(const half8_t*)(wt + col * 32 + kb);
  half8_t hi = *(const half8_t*)(wt + col * 32 + kb + 16);
  return __builtin_shufflevector(lo, hi, 0,1,2,3,4,5,6,7,8,9,10,11,12,13,14,15);
}

__device__ inline float8_t wmma16(half16_t a, half16_t b, float8_t c) {
  return __builtin_amdgcn_wmma_f32_16x16x32_f16(false, a, false, b, (short)0, c,
                                                false, false);
}

__device__ inline float leaky(float v) {
  // branch-free: max/min map to v_dual_max/min + fma (no v_cmp/v_cndmask chain)
  return fmaxf(v, 0.f) + LEAKY_SLOPE * fminf(v, 0.f);
}

// C tile (VGPR r: lanes 0-15 -> M=r, N=lane; lanes 16-31 -> M=8+r) ->
// bias + optional leaky-ReLU -> next layer's LDS A tile (cols >=16 stay 0).
__device__ inline void c_to_A(float8_t c, float bb, bool act,
                              _Float16* A2, int lane) {
  const int n  = lane & 15;
  const int hi = lane >> 4;
#pragma unroll
  for (int r = 0; r < 8; ++r) {
    float v = c[r] + bb;
    if (act) v = leaky(v);
    A2[(r + (hi << 3)) * 32 + n] = (_Float16)v;
  }
}

__device__ inline float atomic_add_f32(float* p, float v) {
  return __hip_atomic_fetch_add(p, v, __ATOMIC_RELAXED, __HIP_MEMORY_SCOPE_AGENT);
}

__device__ inline void zero_lds(_Float16* lds, int count) {
  for (int i = threadIdx.x; i < count; i += TPB) lds[i] = (_Float16)0;
}

// ---- Kernel 0: convert + pad all weights/biases into ws --------------------
__global__ __launch_bounds__(TPB) void k_prep_w(Params P, _Float16* WT, float* BI) {
  for (int idx = threadIdx.x + blockIdx.x * blockDim.x; idx < 18 * 512;
       idx += blockDim.x * gridDim.x) {
    const int j = idx >> 9, e = idx & 511;
    const int col = e >> 5, k = e & 31;
    float v = 0.f;
    if (col < NDIM[j] && k < KDIM[j]) v = P.W[j][k * NDIM[j] + col];  // W is [K,N] row-major
    WT[idx] = (_Float16)v;
  }
  for (int idx = threadIdx.x + blockIdx.x * blockDim.x; idx < 18 * 16;
       idx += blockDim.x * gridDim.x) {
    const int j = idx >> 4, n = idx & 15;
    BI[idx] = (n < NDIM[j]) ? P.B[j][n] : 0.f;
  }
}

// ---- Kernel 1: node_prep  h = MLP(x) ---------------------------------------
__global__ __launch_bounds__(TPB)
void k_node_prep(const float* __restrict__ x, const _Float16* __restrict__ WT,
                 const float* __restrict__ BI, float* __restrict__ hout,
                 int nTiles, int iters) {
  __shared__ _Float16 lds[WPB * 1024];
  zero_lds(lds, WPB * 1024);
  __syncthreads();
  const int wave = threadIdx.x >> 5, lane = threadIdx.x & 31;
  const int n = lane & 15, hi = lane >> 4;
  _Float16* A1 = lds + wave * 1024;
  _Float16* A2 = A1 + 512;
  const half16_t B1 = load_B(WT + 0 * 512, lane);
  const half16_t B2 = load_B(WT + 1 * 512, lane);
  const float bb1 = BI[0 * 16 + n];
  const float bb2 = BI[1 * 16 + n];
  const float8_t z = {};
  for (int it = 0; it < iters; ++it) {
    const int tile  = (it * gridDim.x + blockIdx.x) * WPB + wave;
    const bool valid = tile < nTiles;
    const int base  = tile * 16;
    if (valid && lane < 16) {
      const float* xr = x + (size_t)(base + lane) * 5;
#pragma unroll
      for (int j = 0; j < 5; ++j) A1[lane * 32 + j] = (_Float16)xr[j];
    }
    __syncthreads();
    float8_t c1 = wmma16(load_A_lds(A1, lane), B1, z);
    c_to_A(c1, bb1, true, A2, lane);
    __syncthreads();
    float8_t c2 = wmma16(load_A_lds(A2, lane), B2, z);
    if (valid && n < 8) {
#pragma unroll
      for (int r = 0; r < 8; ++r)
        hout[(size_t)(base + r + hi * 8) * 8 + n] = c2[r] + bb2;
    }
  }
}

// ---- Kernel 2: generic 2-layer 8->16->8 MLP (node_msg) ---------------------
__global__ __launch_bounds__(TPB)
void k_mlp8to8(const float* __restrict__ in, const _Float16* __restrict__ WT,
               const float* __restrict__ BI, int l0, float* __restrict__ out,
               int nTiles, int iters) {
  __shared__ _Float16 lds[WPB * 1024];
  zero_lds(lds, WPB * 1024);
  __syncthreads();
  const int wave = threadIdx.x >> 5, lane = threadIdx.x & 31;
  const int n = lane & 15, hi = lane >> 4;
  _Float16* A1 = lds + wave * 1024;
  _Float16* A2 = A1 + 512;
  const half16_t B1 = load_B(WT + (size_t)l0 * 512, lane);
  const half16_t B2 = load_B(WT + (size_t)(l0 + 1) * 512, lane);
  const float bb1 = BI[l0 * 16 + n];
  const float bb2 = BI[(l0 + 1) * 16 + n];
  const float8_t z = {};
  const int stride = gridDim.x * WPB;
  for (int it = 0; it < iters; ++it) {
    const int tile  = (it * gridDim.x + blockIdx.x) * WPB + wave;
    const bool valid = tile < nTiles;
    const int base  = tile * 16;
    if (valid && lane < 16) {
      const float* ir = in + (size_t)(base + lane) * 8;
      half8_t hv;
#pragma unroll
      for (int j = 0; j < 8; ++j) hv[j] = (_Float16)ir[j];
      *(half8_t*)(A1 + lane * 32) = hv;
      // prefetch next iteration's rows (L2-resident stream)
      const int ntile = tile + stride;
      if (ntile < nTiles)
        __builtin_prefetch(in + (size_t)(ntile * 16 + lane) * 8, 0, 1);
    }
    __syncthreads();
    float8_t c1 = wmma16(load_A_lds(A1, lane), B1, z);
    c_to_A(c1, bb1, true, A2, lane);
    __syncthreads();
    float8_t c2 = wmma16(load_A_lds(A2, lane), B2, z);
    if (valid && n < 8) {
#pragma unroll
      for (int r = 0; r < 8; ++r)
        out[(size_t)(base + r + hi * 8) * 8 + n] = c2[r] + bb2;
    }
  }
}

// ---- Kernel 3: edge scatter  agg[src] += y[dst] ----------------------------
__global__ __launch_bounds__(TPB)
void k_edges(const int* __restrict__ src, const int* __restrict__ dst,
             const float* __restrict__ y, float* __restrict__ agg, int E) {
  const int e = blockIdx.x * blockDim.x + threadIdx.x;
  if (e >= E) return;
  const int s = src[e];
  const int d = dst[e];
  const float4* yp = (const float4*)(y + (size_t)d * 8);
  const float4 a = yp[0], b = yp[1];
  float* ap = agg + (size_t)s * 8;
  atomic_add_f32(ap + 0, a.x); atomic_add_f32(ap + 1, a.y);
  atomic_add_f32(ap + 2, a.z); atomic_add_f32(ap + 3, a.w);
  atomic_add_f32(ap + 4, b.x); atomic_add_f32(ap + 5, b.y);
  atomic_add_f32(ap + 6, b.z); atomic_add_f32(ap + 7, b.w);
}

// ---- Kernel 4: node_update + masked residual  h += mask * MLP(agg) ---------
__global__ __launch_bounds__(TPB)
void k_update(const float* __restrict__ agg, const _Float16* __restrict__ WT,
              const float* __restrict__ BI, const float* __restrict__ mask,
              float* __restrict__ hbuf, int nTiles, int iters) {
  __shared__ _Float16 lds[WPB * 1024];
  zero_lds(lds, WPB * 1024);
  __syncthreads();
  const int wave = threadIdx.x >> 5, lane = threadIdx.x & 31;
  const int n = lane & 15, hi = lane >> 4;
  _Float16* A1 = lds + wave * 1024;
  _Float16* A2 = A1 + 512;
  const half16_t B1 = load_B(WT + 4 * 512, lane);
  const half16_t B2 = load_B(WT + 5 * 512, lane);
  const float bb1 = BI[4 * 16 + n];
  const float bb2 = BI[5 * 16 + n];
  const float8_t z = {};
  const int stride = gridDim.x * WPB;
  for (int it = 0; it < iters; ++it) {
    const int tile  = (it * gridDim.x + blockIdx.x) * WPB + wave;
    const bool valid = tile < nTiles;
    const int base  = tile * 16;
    if (valid && lane < 16) {
      const float* ir = agg + (size_t)(base + lane) * 8;
      half8_t hv;
#pragma unroll
      for (int j = 0; j < 8; ++j) hv[j] = (_Float16)ir[j];
      *(half8_t*)(A1 + lane * 32) = hv;
      const int ntile = tile + stride;
      if (ntile < nTiles)
        __builtin_prefetch(agg + (size_t)(ntile * 16 + lane) * 8, 0, 1);
    }
    __syncthreads();
    float8_t c1 = wmma16(load_A_lds(A1, lane), B1, z);
    c_to_A(c1, bb1, true, A2, lane);
    __syncthreads();
    float8_t c2 = wmma16(load_A_lds(A2, lane), B2, z);
    if (valid && n < 8) {
#pragma unroll
      for (int r = 0; r < 8; ++r) {
        const int row = base + r + hi * 8;
        hbuf[(size_t)row * 8 + n] += mask[row] * (c2[r] + bb2);
      }
    }
  }
}

// ---- Kernel 5: dag_msg (3 layers) + per-DAG column reduction ---------------
__global__ __launch_bounds__(TPB)
void k_dag_msg(const float* __restrict__ x, const float* __restrict__ h,
               const _Float16* __restrict__ WT, const float* __restrict__ BI,
               float* __restrict__ dsumm, int nTiles, int npd, int iters) {
  __shared__ _Float16 lds[WPB * 1536];
  zero_lds(lds, WPB * 1536);
  __syncthreads();
  const int wave = threadIdx.x >> 5, lane = threadIdx.x & 31;
  const int n = lane & 15;
  _Float16* A1 = lds + wave * 1536;
  _Float16* A2 = A1 + 512;
  _Float16* A3 = A1 + 1024;
  const half16_t B1 = load_B(WT + 6 * 512, lane);
  const half16_t B2 = load_B(WT + 7 * 512, lane);
  const half16_t B3 = load_B(WT + 8 * 512, lane);
  const float bb1 = BI[6 * 16 + n];
  const float bb2 = BI[7 * 16 + n];
  const float bb3 = BI[8 * 16 + n];
  const float8_t z = {};
  for (int it = 0; it < iters; ++it) {
    const int tile  = (it * gridDim.x + blockIdx.x) * WPB + wave;
    const bool valid = tile < nTiles;
    const int base  = tile * 16;
    if (valid && lane < 16) {
      const int row = base + lane;
      const float* xr = x + (size_t)row * 5;
      const float* hr = h + (size_t)row * 8;
      _Float16* Ar = A1 + lane * 32;
#pragma unroll
      for (int j = 0; j < 5; ++j) Ar[j] = (_Float16)xr[j];
#pragma unroll
      for (int j = 0; j < 8; ++j) Ar[5 + j] = (_Float16)hr[j];
    }
    __syncthreads();
    float8_t c1 = wmma16(load_A_lds(A1, lane), B1, z);
    c_to_A(c1, bb1, true, A2, lane);
    __syncthreads();
    float8_t c2 = wmma16(load_A_lds(A2, lane), B2, z);
    c_to_A(c2, bb2, true, A3, lane);
    __syncthreads();
    float8_t c3 = wmma16(load_A_lds(A3, lane), B3, z);
    // column sums over 16 rows (two lanes per column), one atomic per column
    float partial = 0.f;
#pragma unroll
    for (int r = 0; r < 8; ++r) partial += c3[r] + bb3;
    partial += __shfl_xor(partial, 16, 32);
    if (valid && lane < 8) {
      const int dag = base / npd;  // tile lies inside one DAG (16 | npd)
      atomic_add_f32(dsumm + (size_t)dag * 8 + lane, partial);
    }
  }
}

// ---- Kernel 6: glob = sum_g MLP(dag_summ[g]) -------------------------------
__global__ __launch_bounds__(TPB)
void k_glob(const float* __restrict__ dsumm, const _Float16* __restrict__ WT,
            const float* __restrict__ BI, float* __restrict__ glob,
            int nTiles, int G) {
  __shared__ _Float16 lds[WPB * 1536];
  zero_lds(lds, WPB * 1536);
  __syncthreads();
  const int wave = threadIdx.x >> 5, lane = threadIdx.x & 31;
  const int n = lane & 15;
  _Float16* A1 = lds + wave * 1536;
  _Float16* A2 = A1 + 512;
  _Float16* A3 = A1 + 1024;
  const half16_t B1 = load_B(WT + 9 * 512, lane);
  const half16_t B2 = load_B(WT + 10 * 512, lane);
  const half16_t B3 = load_B(WT + 11 * 512, lane);
  const float bb1 = BI[9 * 16 + n];
  const float bb2 = BI[10 * 16 + n];
  const float bb3 = BI[11 * 16 + n];
  const int tile  = blockIdx.x * WPB + wave;
  const bool valid = tile < nTiles;
  const int base  = tile * 16;
  if (valid && lane < 16 && (base + lane) < G) {
    const float* ir = dsumm + (size_t)(base + lane) * 8;
    half8_t hv;
#pragma unroll
    for (int j = 0; j < 8; ++j) hv[j] = (_Float16)ir[j];
    *(half8_t*)(A1 + lane * 32) = hv;
  }
  __syncthreads();
  const float8_t z = {};
  float8_t c1 = wmma16(load_A_lds(A1, lane), B1, z);
  c_to_A(c1, bb1, true, A2, lane);
  __syncthreads();
  float8_t c2 = wmma16(load_A_lds(A2, lane), B2, z);
  c_to_A(c2, bb2, true, A3, lane);
  __syncthreads();
  float8_t c3 = wmma16(load_A_lds(A3, lane), B3, z);
  float partial = 0.f;
#pragma unroll
  for (int r = 0; r < 8; ++r) partial += c3[r] + bb3;  // G % 16 == 0 assumed
  partial += __shfl_xor(partial, 16, 32);
  if (valid && lane < 8) atomic_add_f32(glob + lane, partial);
}

// ---- Kernel 7: node scores -------------------------------------------------
__global__ __launch_bounds__(TPB)
void k_node_score(const float* __restrict__ x, const float* __restrict__ h,
                  const float* __restrict__ dsumm, const float* __restrict__ glob,
                  const _Float16* __restrict__ WT, const float* __restrict__ BI,
                  float* __restrict__ out, int nTiles, int npd, int iters) {
  __shared__ _Float16 lds[WPB * 1536];
  zero_lds(lds, WPB * 1536);
  __syncthreads();
  const int wave = threadIdx.x >> 5, lane = threadIdx.x & 31;
  const int n = lane & 15, hi = lane >> 4;
  _Float16* A1 = lds + wave * 1536;
  _Float16* A2 = A1 + 512;
  _Float16* A3 = A1 + 1024;
  const half16_t B1 = load_B(WT + 12 * 512, lane);
  const half16_t B2 = load_B(WT + 13 * 512, lane);
  const half16_t B3 = load_B(WT + 14 * 512, lane);
  const float bb1 = BI[12 * 16 + n];
  const float bb2 = BI[13 * 16 + n];
  const float bb3 = BI[14 * 16 + 0];
  float gl[8];
#pragma unroll
  for (int j = 0; j < 8; ++j) gl[j] = glob[j];
  const float8_t z = {};
  for (int it = 0; it < iters; ++it) {
    const int tile  = (it * gridDim.x + blockIdx.x) * WPB + wave;
    const bool valid = tile < nTiles;
    const int base  = tile * 16;
    if (valid && lane < 16) {
      const int row = base + lane;
      const int dag = row / npd;
      const float* xr = x + (size_t)row * 5;
      const float* hr = h + (size_t)row * 8;
      const float* dr = dsumm + (size_t)dag * 8;
      _Float16* Ar = A1 + lane * 32;
#pragma unroll
      for (int j = 0; j < 5; ++j) Ar[j] = (_Float16)xr[j];
#pragma unroll
      for (int j = 0; j < 8; ++j) Ar[5 + j] = (_Float16)hr[j];
#pragma unroll
      for (int j = 0; j < 8; ++j) Ar[13 + j] = (_Float16)dr[j];
#pragma unroll
      for (int j = 0; j < 8; ++j) Ar[21 + j] = (_Float16)gl[j];
    }
    __syncthreads();
    float8_t c1 = wmma16(load_A_lds(A1, lane), B1, z);
    c_to_A(c1, bb1, true, A2, lane);
    __syncthreads();
    float8_t c2 = wmma16(load_A_lds(A2, lane), B2, z);
    c_to_A(c2, bb2, true, A3, lane);
    __syncthreads();
    float8_t c3 = wmma16(load_A_lds(A3, lane), B3, z);
    if (valid && n == 0) {  // column 0 lives in lanes 0 (M 0-7) and 16 (M 8-15)
#pragma unroll
      for (int r = 0; r < 8; ++r) out[base + r + hi * 8] = c3[r] + bb3;
    }
  }
}

// ---- Kernel 8: dag scores (G*K rows of width 20) ---------------------------
__global__ __launch_bounds__(TPB)
void k_dag_score(const int* __restrict__ ptr, const float* __restrict__ x,
                 const float* __restrict__ dsumm, const float* __restrict__ glob,
                 const _Float16* __restrict__ WT, const float* __restrict__ BI,
                 float* __restrict__ out, int R, int K, int nTiles, int iters) {
  __shared__ _Float16 lds[WPB * 1536];
  zero_lds(lds, WPB * 1536);
  __syncthreads();
  const int wave = threadIdx.x >> 5, lane = threadIdx.x & 31;
  const int n = lane & 15, hi = lane >> 4;
  _Float16* A1 = lds + wave * 1536;
  _Float16* A2 = A1 + 512;
  _Float16* A3 = A1 + 1024;
  const half16_t B1 = load_B(WT + 15 * 512, lane);
  const half16_t B2 = load_B(WT + 16 * 512, lane);
  const half16_t B3 = load_B(WT + 17 * 512, lane);
  const float bb1 = BI[15 * 16 + n];
  const float bb2 = BI[16 * 16 + n];
  const float bb3 = BI[17 * 16 + 0];
  float gl[8];
#pragma unroll
  for (int j = 0; j < 8; ++j) gl[j] = glob[j];
  const float8_t z = {};
  for (int it = 0; it < iters; ++it) {
    const int tile  = (it * gridDim.x + blockIdx.x) * WPB + wave;
    const bool valid = tile < nTiles;
    const int base  = tile * 16;
    if (valid && lane < 16 && (base + lane) < R) {
      const int row = base + lane;
      const int g  = row / K;
      const int kk = row - g * K;
      const int nd = ptr[g];
      const float* xr = x + (size_t)nd * 5;
      const float* dr = dsumm + (size_t)g * 8;
      _Float16* Ar = A1 + lane * 32;
#pragma unroll
      for (int j = 0; j < 3; ++j) Ar[j] = (_Float16)xr[j];
#pragma unroll
      for (int j = 0; j < 8; ++j) Ar[3 + j] = (_Float16)dr[j];
#pragma unroll
      for (int j = 0; j < 8; ++j) Ar[11 + j] = (_Float16)gl[j];
      Ar[19] = (_Float16)(float)kk;
    }
    __syncthreads();
    float8_t c1 = wmma16(load_A_lds(A1, lane), B1, z);
    c_to_A(c1, bb1, true, A2, lane);
    __syncthreads();
    float8_t c2 = wmma16(load_A_lds(A2, lane), B2, z);
    c_to_A(c2, bb2, true, A3, lane);
    __syncthreads();
    float8_t c3 = wmma16(load_A_lds(A3, lane), B3, z);
    if (valid && n == 0) {
#pragma unroll
      for (int r = 0; r < 8; ++r) {
        const int row = base + r + hi * 8;
        if (row < R) out[row] = c3[r] + bb3;
      }
    }
  }
}

// ---------------------------------------------------------------------------
static inline void tile_grid(int tiles, int& blocks, int& iters) {
  blocks = (tiles + WPB - 1) / WPB;
  if (blocks > MAXBLK) blocks = MAXBLK;
  iters = (tiles + blocks * WPB - 1) / (blocks * WPB);
}

extern "C" void kernel_launch(void* const* d_in, const int* in_sizes, int n_in,
                              void* d_out, int out_size, void* d_ws, size_t ws_size,
                              hipStream_t stream) {
  const float* x     = (const float*)d_in[0];
  const int*   ptr   = (const int*)d_in[1];
  const int*   lsrc  = (const int*)d_in[2];
  const int*   ldst  = (const int*)d_in[3];
  const float* lmask = (const float*)d_in[4];

  const int N  = in_sizes[0] / 5;
  const int G  = in_sizes[1] - 1;
  const int Lv = in_sizes[4] / N;
  const int E  = in_sizes[2] / Lv;
  const int K  = (out_size - N) / G;   // derive num_executors without readback
  const int npd = N / G;

  // Workspace map: [0,32K) weight tiles + biases; then f32 buffers (L2-resident).
  char* w = (char*)d_ws;
  _Float16* WT  = (_Float16*)w;                 // 18*512 halves = 18 KB
  float*    BI  = (float*)(w + 24 * 1024);      // 18*16 floats
  float*    hB  = (float*)(w + 32 * 1024);      // N*8
  float*    yB  = hB + (size_t)N * 8;           // N*8
  float*    agg = yB + (size_t)N * 8;           // N*8
  float*    dsm = agg + (size_t)N * 8;          // G*8
  float*    glb = dsm + (size_t)G * 8;          // 8

  Params P;
  for (int j = 0; j < 18; ++j) {
    P.W[j] = (const float*)d_in[5 + 2 * j];
    P.B[j] = (const float*)d_in[6 + 2 * j];
  }

  hipLaunchKernelGGL(k_prep_w, dim3(4), dim3(TPB), 0, stream, P, WT, BI);

  const int tilesN = (N + 15) / 16;
  const int tilesG = (G + 15) / 16;
  const int R      = G * K;
  const int tilesR = (R + 15) / 16;
  int blkN, itN, blkR, itR;
  tile_grid(tilesN, blkN, itN);
  tile_grid(tilesR, blkR, itR);
  const int blkG   = (tilesG + WPB - 1) / WPB;
  const int blocksE = (E + TPB - 1) / TPB;

  hipLaunchKernelGGL(k_node_prep, dim3(blkN), dim3(TPB), 0, stream,
                     x, WT, BI, hB, tilesN, itN);

  for (int l = 0; l < Lv; ++l) {
    hipLaunchKernelGGL(k_mlp8to8, dim3(blkN), dim3(TPB), 0, stream,
                       hB, WT, BI, 2, yB, tilesN, itN);
    hipMemsetAsync(agg, 0, (size_t)N * 8 * sizeof(float), stream);
    hipLaunchKernelGGL(k_edges, dim3(blocksE), dim3(TPB), 0, stream,
                       lsrc + (size_t)l * E, ldst + (size_t)l * E, yB, agg, E);
    hipLaunchKernelGGL(k_update, dim3(blkN), dim3(TPB), 0, stream,
                       agg, WT, BI, lmask + (size_t)l * N, hB, tilesN, itN);
  }

  hipMemsetAsync(dsm, 0, (size_t)G * 8 * sizeof(float), stream);
  hipMemsetAsync(glb, 0, 8 * sizeof(float), stream);

  hipLaunchKernelGGL(k_dag_msg, dim3(blkN), dim3(TPB), 0, stream,
                     x, hB, WT, BI, dsm, tilesN, npd, itN);
  hipLaunchKernelGGL(k_glob, dim3(blkG), dim3(TPB), 0, stream,
                     dsm, WT, BI, glb, tilesG, G);
  hipLaunchKernelGGL(k_node_score, dim3(blkN), dim3(TPB), 0, stream,
                     x, hB, dsm, glb, WT, BI, (float*)d_out, tilesN, npd, itN);
  hipLaunchKernelGGL(k_dag_score, dim3(blkR), dim3(TPB), 0, stream,
                     ptr, x, dsm, glb, WT, BI, (float*)d_out + N, R, K, tilesR, itR);
}
